// DQShift_39290360824521
// MI455X (gfx1250) — matrix operand order repaired and checked by reading, not compiled, not run on hardware
//
#include <hip/hip_runtime.h>
#include <hip/hip_bf16.h>
#include <math.h>

typedef __attribute__((ext_vector_type(16))) _Float16 v16h;
typedef __attribute__((ext_vector_type(8)))  float    v8f;

#define GROUPS  4
#define HH      64
#define WW      64
#define CC      768
#define NN      (HH*WW)
#define TILE    16
#define HALO_W  18          // x0-1 .. x0+16
#define HALO_PIX 54         // 3 rows * 18 cols
#define KSTEPS  (CC/32)     // 24 wmma K-steps
#define C4      (CC/4)      // 192 float4 per pixel

// ---------------------------------------------------------------------------
// Kernel 1: fused depthwise-conv + GELU + dual WMMA projection + offset grid.
// One wave (32 threads) owns a 16-pixel strip of one row of one image.
//   off[b, o, y, x] = sum_c gelu(dwconv(x)[c]) * off_w[o,c] + off_b[o]
//   wt             = sigmoid(sum_c x[c] * wt_w[o,c] + wt_b[o])
//   grid offset    = off*wt + shift  ->  offx/offy workspace (B,GROUPS,N)
// ---------------------------------------------------------------------------
__global__ __launch_bounds__(32)
void dqshift_offsets_kernel(const float* __restrict__ x,     // (B,N,C)
                            const float* __restrict__ dw_w,  // (C,1,3,3)
                            const float* __restrict__ dw_b,  // (C)
                            const float* __restrict__ off_w, // (8,C)
                            const float* __restrict__ off_b, // (8)
                            const float* __restrict__ wt_w,  // (8,C)
                            const float* __restrict__ wt_b,  // (8)
                            float* __restrict__ offx,        // (B,GROUPS,N)
                            float* __restrict__ offy)
{
    extern __shared__ _Float16 halo[];   // [CC][HALO_PIX], stride 54 (odd dwords)

    const int tx   = blockIdx.x;         // 16-pixel tile within row (0..3)
    const int y    = blockIdx.y;         // row (0..63)
    const int b    = blockIdx.z;         // batch
    const int x0   = tx * TILE;
    const int lane = threadIdx.x;        // 0..31
    const int m    = lane & 15;          // A-row (pixel) == B/D column (output)
    const int h    = lane >> 4;          // K-half selector

    // ---- stage 3x18 halo (zero-padded conv border), all 768 channels, f16 ----
    const float4* x4 = (const float4*)x;
    for (int idx = lane; idx < HALO_PIX * C4; idx += 32) {
        int p  = idx / C4;               // halo pixel 0..53
        int c4 = idx - p * C4;
        int r  = p / HALO_W;
        int cc = p - r * HALO_W;
        int yy = y - 1 + r;
        int xx = x0 - 1 + cc;
        float4 v = make_float4(0.f, 0.f, 0.f, 0.f);
        if (yy >= 0 && yy < HH && xx >= 0 && xx < WW)
            v = x4[((size_t)b * NN + (size_t)yy * WW + xx) * C4 + c4];
        int ch = c4 * 4;
        halo[(ch + 0) * HALO_PIX + p] = (_Float16)v.x;
        halo[(ch + 1) * HALO_PIX + p] = (_Float16)v.y;
        halo[(ch + 2) * HALO_PIX + p] = (_Float16)v.z;
        halo[(ch + 3) * HALO_PIX + p] = (_Float16)v.w;
    }
    __syncthreads();

    // ---- dual GEMM over K=768: D_off = gelu(conv(x)) @ off_w^T,
    //                            D_wt  = x @ wt_w^T  (both 16x16 tiles) ----
    v8f c_off = {};
    v8f c_wt  = {};
    for (int ks = 0; ks < KSTEPS; ++ks) {
        const int k0 = ks * 32;

        // A fragments. A 16x32 in contiguous k order used here (k = 16h + i): lanes 0-15 carry K {0..15},
        // lanes 16-31 carry K {16..31} of row m.
        v16h a_off, a_wt;
        #pragma unroll
        for (int i = 0; i < 16; ++i) {
            const int ch = k0 + 16 * h + i;
            const _Float16* hp = &halo[ch * HALO_PIX];
            const float*    wp = &dw_w[ch * 9];
            float acc = dw_b[ch];
            #pragma unroll
            for (int dy = 0; dy < 3; ++dy)
                #pragma unroll
                for (int dx = 0; dx < 3; ++dx)
                    acc += wp[dy * 3 + dx] * (float)hp[dy * HALO_W + m + dx];
            float gel = 0.5f * acc * (1.0f + erff(acc * 0.70710678118654752f));
            a_off[i] = (_Float16)gel;
            a_wt[i]  = hp[HALO_W + m + 1];          // center pixel, channel ch
        }

        // B fragments. B 32x16, same k order as A: lane = N + 16*Khalf, 16 contiguous K per
        // lane -> straight contiguous loads from (O,C) row-major weights.
        v16h b_off = {};
        v16h b_wt  = {};
        if (m < 8) {
            const float* po = &off_w[m * CC + k0 + 16 * h];
            const float* pw = &wt_w [m * CC + k0 + 16 * h];
            #pragma unroll
            for (int i = 0; i < 16; ++i) {
                b_off[i] = (_Float16)po[i];
                b_wt[i]  = (_Float16)pw[i];
            }
        }

        c_off = __builtin_amdgcn_wmma_f32_16x16x32_f16(false, a_off, false, b_off,
                                                       (short)0, c_off, false, false);
        c_wt  = __builtin_amdgcn_wmma_f32_16x16x32_f16(false, a_wt,  false, b_wt,
                                                       (short)0, c_wt,  false, false);
    }

    // ---- epilogue: D layout lane = column o, VGPR j = row (pixel) j + 8*h ----
    const int o = m;                      // output channel 0..7 valid
    if (o < 8) {
        const float shx[4] = { 0.f, 0.f, -1.f, 1.f };
        const float shy[4] = { -1.f, 1.f, 0.f, 0.f };
        const int   g   = o & 3;
        const int   dim = o >> 2;
        const float ob  = off_b[o];
        const float wb  = wt_b[o];
        const float sh  = dim ? shy[g] : shx[g];
        float* dst = dim ? offy : offx;
        #pragma unroll
        for (int j = 0; j < 8; ++j) {
            const int px  = x0 + j + 8 * h;
            const float ov  = c_off[j] + ob;
            const float sig = 1.0f / (1.0f + __expf(-(c_wt[j] + wb)));
            dst[((size_t)(b * GROUPS + g)) * NN + (size_t)y * WW + px] = ov * sig + sh;
        }
    }
}

// ---------------------------------------------------------------------------
// Kernel 2: bilinear grid-sample with border clamp, 4 channels per thread.
// Reference's normalize/denormalize collapses exactly to:
//   ix = clamp(px + offx, 0, W-1), iy = clamp(py + offy, 0, H-1)
// float4 gathers/stores: fully coalesced b128 traffic; the 4 neighborhoods of
// each pixel get ~4x reuse out of the 192MB L2 (whole 96MB input resident).
// ---------------------------------------------------------------------------
__global__ __launch_bounds__(256)
void dqshift_sample_kernel(const float* __restrict__ x,
                           const float* __restrict__ offx,
                           const float* __restrict__ offy,
                           float* __restrict__ out, int total4)
{
    const int id = blockIdx.x * 256 + threadIdx.x;
    if (id >= total4) return;
    const int c4   = id % C4;            // float4 index within pixel (0..191)
    const int rest = id / C4;
    const int n    = rest & (NN - 1);
    const int b    = rest >> 12;
    const int g    = c4 / (C4 / GROUPS); // 48 float4 per group -> uniform per thread
    const int px   = n & (WW - 1);
    const int py   = n >> 6;

    const size_t gi = (size_t)(b * GROUPS + g) * NN + n;
    __builtin_prefetch(&offx[gi + WW], 0, 1);   // next row's offsets -> L2
    const float ix = fminf(fmaxf((float)px + offx[gi], 0.f), (float)(WW - 1));
    const float iy = fminf(fmaxf((float)py + offy[gi], 0.f), (float)(HH - 1));
    const float fx0 = floorf(ix), fy0 = floorf(iy);
    const float wx = ix - fx0,   wy = iy - fy0;
    const int xi0 = (int)fx0, yi0 = (int)fy0;
    const int xi1 = min(xi0 + 1, WW - 1), yi1 = min(yi0 + 1, HH - 1);

    const float w00 = (1.f - wx) * (1.f - wy);
    const float w01 = wx * (1.f - wy);
    const float w10 = (1.f - wx) * wy;
    const float w11 = wx * wy;

    const float4* xb = (const float4*)x + (size_t)b * NN * C4;
    const float4 v00 = xb[(size_t)(yi0 * WW + xi0) * C4 + c4];
    const float4 v01 = xb[(size_t)(yi0 * WW + xi1) * C4 + c4];
    const float4 v10 = xb[(size_t)(yi1 * WW + xi0) * C4 + c4];
    const float4 v11 = xb[(size_t)(yi1 * WW + xi1) * C4 + c4];

    float4 r;
    r.x = v00.x * w00 + v01.x * w01 + v10.x * w10 + v11.x * w11;
    r.y = v00.y * w00 + v01.y * w01 + v10.y * w10 + v11.y * w11;
    r.z = v00.z * w00 + v01.z * w01 + v10.z * w10 + v11.z * w11;
    r.w = v00.w * w00 + v01.w * w01 + v10.w * w10 + v11.w * w11;
    ((float4*)out)[id] = r;
}

extern "C" void kernel_launch(void* const* d_in, const int* in_sizes, int n_in,
                              void* d_out, int out_size, void* d_ws, size_t ws_size,
                              hipStream_t stream) {
    const float* x     = (const float*)d_in[0];
    const float* dw_w  = (const float*)d_in[1];
    const float* dw_b  = (const float*)d_in[2];
    const float* off_w = (const float*)d_in[3];
    const float* off_b = (const float*)d_in[4];
    const float* wt_w  = (const float*)d_in[5];
    const float* wt_b  = (const float*)d_in[6];
    float* out = (float*)d_out;

    const int B = in_sizes[0] / (NN * CC);

    float* offx = (float*)d_ws;                 // B*GROUPS*NN floats
    float* offy = offx + (size_t)B * GROUPS * NN;

    const size_t smem = (size_t)CC * HALO_PIX * sizeof(_Float16);  // ~81 KB
    dim3 grid1(WW / TILE, HH, B);
    dqshift_offsets_kernel<<<grid1, 32, smem, stream>>>(
        x, dw_w, dw_b, off_w, off_b, wt_w, wt_b, offx, offy);

    const int total4 = B * NN * C4;
    dqshift_sample_kernel<<<(total4 + 255) / 256, 256, 0, stream>>>(
        x, offx, offy, out, total4);
}
